// CausalSelfAttention_42245298323571
// MI455X (gfx1250) — matrix operand order repaired and checked by reading, not compiled, not run on hardware
//
#include <hip/hip_runtime.h>
#include <stdint.h>
#include <stddef.h>

// ---------------- problem constants ----------------
#define D_MODEL   1024
#define N_HEADS   16
#define HEAD_DIM  64
#define SEQ       2048
#define BATCH     2
#define M_TOTAL   (BATCH * SEQ)    // 4096 rows of x
#define N_QKV     (3 * D_MODEL)    // 3072

// ---------------- vector types ----------------
typedef __attribute__((ext_vector_type(16))) __bf16   bf16x16;
typedef __attribute__((ext_vector_type(8)))  __bf16   bf16x8;
typedef __attribute__((ext_vector_type(8)))  float    f32x8;
typedef __attribute__((ext_vector_type(4)))  unsigned u32x4;

// ---------------- helpers ----------------
__device__ inline unsigned short f32_to_bf16_bits(float f) {
  unsigned u = __builtin_bit_cast(unsigned, f);
  unsigned r = u + 0x7FFFu + ((u >> 16) & 1u);   // round-to-nearest-even
  return (unsigned short)(r >> 16);
}
__device__ inline __bf16 to_bf16(float f) {
  unsigned short h = f32_to_bf16_bits(f);
  return __builtin_bit_cast(__bf16, h);
}
__device__ inline f32x8 zero8() {
  f32x8 z = {0.f, 0.f, 0.f, 0.f, 0.f, 0.f, 0.f, 0.f};
  return z;
}

// A-operand fragment (16x32 bf16, M rows x K cols), row-major source.
// Contiguous-k order used here: lane l holds row M = l%16; half = l/16 selects K 0-15 vs 16-31.
__device__ inline bf16x16 load_fragA(const __bf16* base, int ld, int row0, int k0,
                                     int lo16, int half) {
  const __bf16* p = base + (size_t)(row0 + lo16) * ld + (k0 + half * 16);
  bf16x8 lo = *(const bf16x8*)p;
  bf16x8 hi = *(const bf16x8*)(p + 8);
  return __builtin_shufflevector(lo, hi, 0, 1, 2, 3, 4, 5, 6, 7,
                                 8, 9, 10, 11, 12, 13, 14, 15);
}

// B-operand fragment (32x16 bf16, K rows x N cols) from TRANSPOSED storage wT[n][k].
// Contiguous-k order used here: lane l holds column N = l%16; lanes 0-15 K=0..15, lanes 16-31 K=16..31.
__device__ inline bf16x16 load_fragB(const __bf16* baseT, int ld, int col0, int k0,
                                     int lo16, int half) {
  const __bf16* p = baseT + (size_t)(col0 + lo16) * ld + (k0 + half * 16);
  bf16x8 lo = *(const bf16x8*)p;
  bf16x8 hi = *(const bf16x8*)(p + 8);
  return __builtin_shufflevector(lo, hi, 0, 1, 2, 3, 4, 5, 6, 7,
                                 8, 9, 10, 11, 12, 13, 14, 15);
}

__device__ inline f32x8 wmma_bf16(bf16x16 a, bf16x16 b, f32x8 c) {
  return __builtin_amdgcn_wmma_f32_16x16x32_bf16(false, a, false, b,
                                                 (short)0, c, false, false);
}

// ---------------- fp32 -> bf16 convert ----------------
__global__ void __launch_bounds__(256) cvt_kernel(const float* __restrict__ in,
                                                  __bf16* __restrict__ out, int n) {
  int i = blockIdx.x * 256 + threadIdx.x;
  if (i < n) out[i] = to_bf16(in[i]);
}

// out[n*K + k] = bf16(in[k*N + n])  (transpose K x N -> N x K)
__global__ void __launch_bounds__(256) cvtT_kernel(const float* __restrict__ in,
                                                   __bf16* __restrict__ out,
                                                   int K, int N) {
  int i = blockIdx.x * 256 + threadIdx.x;
  if (i < K * N) {
    int n = i / K;
    int k = i - n * K;
    out[i] = to_bf16(in[(size_t)k * N + n]);
  }
}

// ---------------- QKV GEMM: [4096,1024] x [1024,3072] + bias ----------------
// Block = 8 waves, block tile 64(M) x 256(N); wave tile 32 x 64 = 2x4 WMMA tiles.
// Epilogue scatters to q[bh][t][d] (pre-scaled 1/8), k[bh][t][d], vT[bh][d][t].
__global__ void __launch_bounds__(256) qkv_gemm_kernel(
    const __bf16* __restrict__ xb, const __bf16* __restrict__ wT,
    const float* __restrict__ bias,
    __bf16* __restrict__ qb, __bf16* __restrict__ kb, __bf16* __restrict__ vtb) {
  const int lane = threadIdx.x & 31;
  const int lo16 = lane & 15, half = lane >> 4;
  const int wave = threadIdx.x >> 5;
  const int wm = wave & 1, wn = wave >> 1;
  const int bm = blockIdx.x & 63;   // 64 M-blocks
  const int bn = blockIdx.x >> 6;   // 12 N-blocks
  const int m0 = bm * 64 + wm * 32;
  const int n0 = bn * 256 + wn * 64;

  f32x8 acc[2][4];
#pragma unroll
  for (int i = 0; i < 2; i++)
#pragma unroll
    for (int j = 0; j < 4; j++) acc[i][j] = zero8();

  for (int k0 = 0; k0 < D_MODEL; k0 += 32) {
    bf16x16 a0 = load_fragA(xb, D_MODEL, m0, k0, lo16, half);
    bf16x16 a1 = load_fragA(xb, D_MODEL, m0 + 16, k0, lo16, half);
    bf16x16 b0 = load_fragB(wT, D_MODEL, n0, k0, lo16, half);
    bf16x16 b1 = load_fragB(wT, D_MODEL, n0 + 16, k0, lo16, half);
    bf16x16 b2 = load_fragB(wT, D_MODEL, n0 + 32, k0, lo16, half);
    bf16x16 b3 = load_fragB(wT, D_MODEL, n0 + 48, k0, lo16, half);
    acc[0][0] = wmma_bf16(a0, b0, acc[0][0]);
    acc[0][1] = wmma_bf16(a0, b1, acc[0][1]);
    acc[0][2] = wmma_bf16(a0, b2, acc[0][2]);
    acc[0][3] = wmma_bf16(a0, b3, acc[0][3]);
    acc[1][0] = wmma_bf16(a1, b0, acc[1][0]);
    acc[1][1] = wmma_bf16(a1, b1, acc[1][1]);
    acc[1][2] = wmma_bf16(a1, b2, acc[1][2]);
    acc[1][3] = wmma_bf16(a1, b3, acc[1][3]);
  }

#pragma unroll
  for (int j = 0; j < 4; j++) {
    const int n = n0 + j * 16 + lo16;      // global output column (0..3071)
    const float bv = bias[n];
    const int which = n >> 10;             // 0=q, 1=k, 2=v
    const int cc = n & 1023;
    const int h = cc >> 6;                 // head
    const int d = cc & 63;                 // dim in head
#pragma unroll
    for (int i = 0; i < 2; i++) {
#pragma unroll
      for (int r = 0; r < 8; r++) {
        const int m = m0 + i * 16 + r + 8 * half;  // global row of x
        const int b = m >> 11;                     // batch
        const int t = m & 2047;                    // seq position
        const int bh = b * N_HEADS + h;
        float v = acc[i][j][r] + bv;
        if (which == 0) {
          qb[((size_t)bh * SEQ + t) * HEAD_DIM + d] = to_bf16(v * 0.125f);
        } else if (which == 1) {
          kb[((size_t)bh * SEQ + t) * HEAD_DIM + d] = to_bf16(v);
        } else {
          vtb[((size_t)bh * HEAD_DIM + d) * SEQ + t] = to_bf16(v);
        }
      }
    }
  }
}

// ---------------- Flash attention (causal, streaming softmax) ----------------
// One wave per 16-row Q tile of one (b,h). Computes S^T = K·Q^T so softmax over
// keys is within-lane; O^T = V^T·P^T accumulated in 4 f32x8 tiles (16x64 output).
__global__ void __launch_bounds__(128) attn_kernel(
    const __bf16* __restrict__ qb, const __bf16* __restrict__ kb,
    const __bf16* __restrict__ vtb, __bf16* __restrict__ ao) {
  const int lane = threadIdx.x & 31;
  const int lo16 = lane & 15, half = lane >> 4;
  const int wave = threadIdx.x >> 5;
  const int flat = blockIdx.x * 4 + wave;   // 0..4095
  const int bh = flat >> 7;                 // (b*16+h)
  const int t0 = (flat & 127) << 4;         // Q tile base row

  const __bf16* qh = qb + (size_t)bh * SEQ * HEAD_DIM;
  const __bf16* kh = kb + (size_t)bh * SEQ * HEAD_DIM;
  const __bf16* vh = vtb + (size_t)bh * HEAD_DIM * SEQ;

  // Q^T B-operand fragments: column t = lane, K = head dim (2 chunks of 32)
  bf16x16 bq0 = load_fragB(qh, HEAD_DIM, t0, 0, lo16, half);
  bf16x16 bq1 = load_fragB(qh, HEAD_DIM, t0, 32, lo16, half);

  f32x8 o[4];
#pragma unroll
  for (int dt = 0; dt < 4; dt++) o[dt] = zero8();
  float mrun = -1e30f, lrun = 0.0f;
  const int myt = t0 + lo16;               // this lane's query index (column of S^T)

  for (int s0 = 0; s0 <= t0; s0 += 32) {
    // --- S^T tiles (2 x 16 keys) ---
    f32x8 st[2];
#pragma unroll
    for (int u = 0; u < 2; u++) {
      const int sb = s0 + u * 16;
      bf16x16 ak0 = load_fragA(kh, HEAD_DIM, sb, 0, lo16, half);
      bf16x16 ak1 = load_fragA(kh, HEAD_DIM, sb, 32, lo16, half);
      f32x8 c = zero8();
      c = wmma_bf16(ak0, bq0, c);
      c = wmma_bf16(ak1, bq1, c);
      st[u] = c;
    }
    // --- causal mask + running max ---
    float pmax = -1e30f;
#pragma unroll
    for (int u = 0; u < 2; u++) {
#pragma unroll
      for (int r = 0; r < 8; r++) {
        const int s = s0 + u * 16 + r + 8 * half;  // key index of this element
        float v = (s <= myt) ? st[u][r] : -1e30f;
        st[u][r] = v;
        pmax = fmaxf(pmax, v);
      }
    }
    pmax = fmaxf(pmax, __shfl_xor(pmax, 16, 32));  // merge lane halves (same column)
    const float mnew = fmaxf(mrun, pmax);
    const float resc = __expf(mrun - mnew);
    mrun = mnew;
    lrun *= resc;
#pragma unroll
    for (int dt = 0; dt < 4; dt++)
#pragma unroll
      for (int r = 0; r < 8; r++) o[dt][r] *= resc;

    // --- P = exp(S - m), running sum ---
    float p[2][8];
    float psum = 0.f;
#pragma unroll
    for (int u = 0; u < 2; u++) {
#pragma unroll
      for (int r = 0; r < 8; r++) {
        float e = __expf(st[u][r] - mnew);
        p[u][r] = e;
        psum += e;
      }
    }
    psum += __shfl_xor(psum, 16, 32);
    lrun += psum;

    // --- pack P^T into B layout: lane half0 needs s=0..15, half1 s=16..31 ---
    bf16x16 bp;
#pragma unroll
    for (int r = 0; r < 8; r++) {
      float q0 = __shfl_xor(p[0][r], 16, 32);  // partner tile0 (s=8..15 for half0)
      float q1 = __shfl_xor(p[1][r], 16, 32);  // partner tile1 (s=16..23 for half1)
      float elo = half ? q1 : p[0][r];
      float ehi = half ? p[1][r] : q0;
      bp[r] = to_bf16(elo);
      bp[8 + r] = to_bf16(ehi);
    }

    // --- O^T += V^T(dtile, s-chunk) * P^T ---
#pragma unroll
    for (int dt = 0; dt < 4; dt++) {
      bf16x16 av = load_fragA(vh, SEQ, dt * 16, s0, lo16, half);
      o[dt] = wmma_bf16(av, bp, o[dt]);
    }
  }

  // --- normalize + store bf16 (packed 16B per dtile) ---
  const float inv = 1.0f / lrun;
  const int b = bh >> 4, h = bh & 15;
  __bf16* dst = ao + ((size_t)(b * SEQ + myt)) * D_MODEL + h * HEAD_DIM + half * 8;
#pragma unroll
  for (int dt = 0; dt < 4; dt++) {
    u32x4 pk;
#pragma unroll
    for (int r = 0; r < 4; r++) {
      unsigned lobits = f32_to_bf16_bits(o[dt][2 * r] * inv);
      unsigned hibits = f32_to_bf16_bits(o[dt][2 * r + 1] * inv);
      pk[r] = lobits | (hibits << 16);
    }
    *(u32x4*)(dst + dt * 16) = pk;
  }
}

// ---------------- Projection GEMM: [4096,1024] x [1024,1024] + bias -> fp32 ----
__global__ void __launch_bounds__(256) proj_gemm_kernel(
    const __bf16* __restrict__ ao, const __bf16* __restrict__ wT,
    const float* __restrict__ bias, float* __restrict__ out) {
  const int lane = threadIdx.x & 31;
  const int lo16 = lane & 15, half = lane >> 4;
  const int wave = threadIdx.x >> 5;
  const int wm = wave & 1, wn = wave >> 1;
  const int bm = blockIdx.x & 63;   // 64 M-blocks
  const int bn = blockIdx.x >> 6;   // 4 N-blocks
  const int m0 = bm * 64 + wm * 32;
  const int n0 = bn * 256 + wn * 64;

  f32x8 acc[2][4];
#pragma unroll
  for (int i = 0; i < 2; i++)
#pragma unroll
    for (int j = 0; j < 4; j++) acc[i][j] = zero8();

  for (int k0 = 0; k0 < D_MODEL; k0 += 32) {
    bf16x16 a0 = load_fragA(ao, D_MODEL, m0, k0, lo16, half);
    bf16x16 a1 = load_fragA(ao, D_MODEL, m0 + 16, k0, lo16, half);
    bf16x16 b0 = load_fragB(wT, D_MODEL, n0, k0, lo16, half);
    bf16x16 b1 = load_fragB(wT, D_MODEL, n0 + 16, k0, lo16, half);
    bf16x16 b2 = load_fragB(wT, D_MODEL, n0 + 32, k0, lo16, half);
    bf16x16 b3 = load_fragB(wT, D_MODEL, n0 + 48, k0, lo16, half);
    acc[0][0] = wmma_bf16(a0, b0, acc[0][0]);
    acc[0][1] = wmma_bf16(a0, b1, acc[0][1]);
    acc[0][2] = wmma_bf16(a0, b2, acc[0][2]);
    acc[0][3] = wmma_bf16(a0, b3, acc[0][3]);
    acc[1][0] = wmma_bf16(a1, b0, acc[1][0]);
    acc[1][1] = wmma_bf16(a1, b1, acc[1][1]);
    acc[1][2] = wmma_bf16(a1, b2, acc[1][2]);
    acc[1][3] = wmma_bf16(a1, b3, acc[1][3]);
  }

#pragma unroll
  for (int j = 0; j < 4; j++) {
    const int n = n0 + j * 16 + lo16;
    const float bv = bias[n];
#pragma unroll
    for (int i = 0; i < 2; i++) {
#pragma unroll
      for (int r = 0; r < 8; r++) {
        const int m = m0 + i * 16 + r + 8 * half;
        out[(size_t)m * D_MODEL + n] = acc[i][j][r] + bv;
      }
    }
  }
}

// ---------------- launcher ----------------
extern "C" void kernel_launch(void* const* d_in, const int* in_sizes, int n_in,
                              void* d_out, int out_size, void* d_ws, size_t ws_size,
                              hipStream_t stream) {
  const float* x      = (const float*)d_in[0];  // [2,2048,1024]
  const float* w_qkv  = (const float*)d_in[1];  // [1024,3072]
  const float* b_qkv  = (const float*)d_in[2];  // [3072]
  const float* w_proj = (const float*)d_in[3];  // [1024,1024]
  const float* b_proj = (const float*)d_in[4];  // [1024]
  float* out = (float*)d_out;

  char* ws = (char*)d_ws;
  const size_t MB = 1024ull * 1024ull;
  __bf16* xb    = (__bf16*)(ws + 0 * MB);   // [4096,1024]        8 MB
  __bf16* wqkvT = (__bf16*)(ws + 8 * MB);   // [3072,1024]        6 MB
  __bf16* wprjT = (__bf16*)(ws + 14 * MB);  // [1024,1024]        2 MB
  __bf16* qb    = (__bf16*)(ws + 16 * MB);  // [32][2048][64]     8 MB (pre-scaled 1/8)
  __bf16* kb    = (__bf16*)(ws + 24 * MB);  // [32][2048][64]     8 MB
  __bf16* vtb   = (__bf16*)(ws + 32 * MB);  // [32][64][2048]     8 MB (transposed V)
  __bf16* ao    = (__bf16*)(ws + 40 * MB);  // [4096,1024]        8 MB

  // 1) converts / transposes to bf16
  {
    int n = M_TOTAL * D_MODEL;
    cvt_kernel<<<(n + 255) / 256, 256, 0, stream>>>(x, xb, n);
  }
  {
    int n = D_MODEL * N_QKV;
    cvtT_kernel<<<(n + 255) / 256, 256, 0, stream>>>(w_qkv, wqkvT, D_MODEL, N_QKV);
  }
  {
    int n = D_MODEL * D_MODEL;
    cvtT_kernel<<<(n + 255) / 256, 256, 0, stream>>>(w_proj, wprjT, D_MODEL, D_MODEL);
  }

  // 2) fused QKV projection (WMMA), scatter to per-head q/k/v^T
  qkv_gemm_kernel<<<64 * (N_QKV / 256), 256, 0, stream>>>(xb, wqkvT, b_qkv,
                                                          qb, kb, vtb);

  // 3) causal flash attention (WMMA), 4096 wave-tiles / 4 waves per block
  attn_kernel<<<1024, 128, 0, stream>>>(qb, kb, vtb, ao);

  // 4) output projection (WMMA) with fp32 epilogue + bias
  proj_gemm_kernel<<<64 * (D_MODEL / 256), 256, 0, stream>>>(ao, wprjT, b_proj, out);

  (void)in_sizes; (void)n_in; (void)out_size; (void)ws_size;
}
